// TorchJSDLoss_25941602467891
// MI455X (gfx1250) — matrix operand order repaired and checked by reading, not compiled, not run on hardware
//
#include <hip/hip_runtime.h>

// ---------------- problem constants (match reference) ----------------
constexpr int    NTOK        = 2048;
constexpr int    HS          = 2048;   // student hidden
constexpr int    HT          = 4096;   // teacher hidden
constexpr int    VOCAB       = 50257;
constexpr int    IGNORE_IDX  = -100;
constexpr float  W_HARD      = 0.5f;
constexpr float  W_SOFT      = 0.5f;
constexpr float  BETA        = 0.5f;
// TEMPERATURE == 1.0 -> no scaling needed

// ---------------- vector types for WMMA ----------------
typedef __attribute__((ext_vector_type(8)))  __bf16 v8bf;
typedef __attribute__((ext_vector_type(16))) __bf16 v16bf;
typedef __attribute__((ext_vector_type(8)))  float  v8f;

union BF16x16 { v16bf v; v8bf h[2]; };

// =====================================================================
// GEMM: out[n, v] = sum_k X[n, k] * W[v, k]   (bf16 -> f32 accum)
// X: [NTOK, K] bf16 row-major; W: [VOCAB, K] bf16 row-major (torch layout)
// -> both operands are K-major, so A and B fragments load with plain
//    contiguous b128s (no transpose path needed).
//
// Workgroup: 256 thr = 8 waves; block tile 256(M) x 128(V)
//   (BM=256 halves HBM weight traffic vs BM=128: W re-read N/BM = 8x).
// Waves arranged 4(M) x 2(V); each wave owns 64x64 = 4x4 subtiles of
// 16x16x32 -> 16 v8f accumulators, 16 WMMA per 32-deep K step.
// =====================================================================
__global__ __launch_bounds__(256) void gemm_logits_kernel(
    const __bf16* __restrict__ X, const __bf16* __restrict__ W,
    float* __restrict__ out, int K)
{
    const int lane  = threadIdx.x & 31;
    const int wave  = threadIdx.x >> 5;
    const int wm    = wave & 3;          // 0..3 -> 64-row slice of 256-row block
    const int wn    = wave >> 2;         // 0..1 -> 64-col slice of 128-col block
    const int l16   = lane & 15;
    const int lhalf = lane >> 4;         // 0 or 1

    const int rowBase = blockIdx.y * 256 + wm * 64;   // NTOK % 256 == 0, no guard
    const int colBase = blockIdx.x * 128 + wn * 64;   // may exceed VOCAB (guarded)

    // A-fragment row pointers (lane holds row M = l16 of each 16-row subtile)
    const __bf16* arow[4];
#pragma unroll
    for (int i = 0; i < 4; ++i)
        arow[i] = X + (size_t)(rowBase + 16 * i + l16) * K;

    // B-fragment row pointers (lane holds column V = l16; clamp for tail tile)
    const __bf16* brow[4];
#pragma unroll
    for (int j = 0; j < 4; ++j) {
        int vcol = colBase + 16 * j + l16;
        if (vcol >= VOCAB) vcol = VOCAB - 1;          // padded cols: garbage, masked at store
        brow[j] = W + (size_t)vcol * K;
    }

    v8f acc[4][4];
#pragma unroll
    for (int i = 0; i < 4; ++i)
#pragma unroll
        for (int j = 0; j < 4; ++j)
            acc[i][j] = (v8f){0.f, 0.f, 0.f, 0.f, 0.f, 0.f, 0.f, 0.f};

    const int aoff = 16 * lhalf;   // A: lanes 0-15 -> K 0..15; lanes 16-31 -> K 16..31
    const int boff = 16 * lhalf;   // B: lanes 0-15 -> K 0..15; lanes 16-31 -> K 16..31

#pragma unroll 1
    for (int kk = 0; kk < K; kk += 32) {
        BF16x16 a[4], b[4];
#pragma unroll
        for (int i = 0; i < 4; ++i) {
            a[i].h[0] = *(const v8bf*)(arow[i] + kk + aoff);
            a[i].h[1] = *(const v8bf*)(arow[i] + kk + aoff + 8);
        }
#pragma unroll
        for (int j = 0; j < 4; ++j)
            b[j].v = *(const v16bf*)(brow[j] + kk + boff);

        // nudge the streaming weight rows ahead along K
        __builtin_prefetch(brow[0] + kk + 128, 0, 0);
        __builtin_prefetch(brow[1] + kk + 128, 0, 0);
        __builtin_prefetch(brow[2] + kk + 128, 0, 0);
        __builtin_prefetch(brow[3] + kk + 128, 0, 0);

#pragma unroll
        for (int i = 0; i < 4; ++i)
#pragma unroll
            for (int j = 0; j < 4; ++j)
                acc[i][j] = __builtin_amdgcn_wmma_f32_16x16x32_bf16(
                    false, a[i].v, false, b[j].v, (short)0, acc[i][j], false, false);
    }

    // C/D layout: lane holds column n = l16; VGPR r holds row (r + 8*lhalf)
#pragma unroll
    for (int i = 0; i < 4; ++i) {
#pragma unroll
        for (int j = 0; j < 4; ++j) {
            const int col = colBase + 16 * j + l16;
            if (col < VOCAB) {
                float* o = out + (size_t)(rowBase + 16 * i + 8 * lhalf) * VOCAB + col;
#pragma unroll
                for (int r = 0; r < 8; ++r)
                    o[(size_t)r * VOCAB] = acc[i][j][r];
            }
        }
    }
}

// =====================================================================
// Per-row loss partials. One 256-thread block per token row.
// Pass 1: ONLINE logsumexp for student & teacher (single sweep).
// Pass 2: JSD accumulation. (2 sweeps of 201KB/row instead of 3.)
// =====================================================================
__global__ __launch_bounds__(256) void row_loss_kernel(
    const float* __restrict__ S, const float* __restrict__ T,
    const int* __restrict__ tgt,
    float* __restrict__ hardP, float* __restrict__ softP)
{
    __shared__ float redm[256];
    __shared__ float reds[256];
    const int n = blockIdx.x;
    const int t = threadIdx.x;
    const float* srow = S + (size_t)n * VOCAB;
    const float* trow = T + (size_t)n * VOCAB;

    // combine (max, sumexp) pairs across the block
    auto blockLse = [&](float m, float s, float& mo, float& so) {
        redm[t] = m; reds[t] = s; __syncthreads();
        for (int w = 128; w > 0; w >>= 1) {
            if (t < w) {
                float m1 = redm[t], s1 = reds[t];
                float m2 = redm[t + w], s2 = reds[t + w];
                float nm = fmaxf(m1, m2);
                redm[t] = nm;
                reds[t] = s1 * __expf(m1 - nm) + s2 * __expf(m2 - nm);
            }
            __syncthreads();
        }
        mo = redm[0]; so = reds[0]; __syncthreads();
    };
    auto blockSum = [&](float v) -> float {
        reds[t] = v; __syncthreads();
        for (int w = 128; w > 0; w >>= 1) {
            if (t < w) reds[t] += reds[t + w];
            __syncthreads();
        }
        float r = reds[0]; __syncthreads(); return r;
    };

    // ---- pass 1: online max + sum-of-exp, both tensors in one sweep ----
    float sm = -3.4e38f, ssum = 0.f;
    float tm = -3.4e38f, tsum = 0.f;
    for (int v = t; v < VOCAB; v += 256) {
        const float x = srow[v];
        const float y = trow[v];
        float nm = fmaxf(sm, x);
        ssum = ssum * __expf(sm - nm) + __expf(x - nm);
        sm = nm;
        nm = fmaxf(tm, y);
        tsum = tsum * __expf(tm - nm) + __expf(y - nm);
        tm = nm;
    }
    float smax, stot, tmax, ttot;
    blockLse(sm, ssum, smax, stot);
    blockLse(tm, tsum, tmax, ttot);
    const float slse = logf(stot);
    const float tlse = logf(ttot);

    // ---- pass 2: JSD accumulation ----
    float kls = 0.f, klt = 0.f;
    for (int v = t; v < VOCAB; v += 256) {
        const float slp = srow[v] - smax - slse;
        const float tlp = trow[v] - tmax - tlse;
        const float sp  = __expf(slp);
        const float tp  = __expf(tlp);
        const float lm  = logf(BETA * sp + (1.f - BETA) * tp);
        kls += sp * (slp - lm);
        klt += tp * (tlp - lm);
    }
    kls = blockSum(kls);
    klt = blockSum(klt);

    if (t == 0) {
        const int  tg    = tgt[n];
        const bool valid = (tg != IGNORE_IDX);
        float ce = 0.f;
        if (valid) {
            int c = tg < 0 ? 0 : (tg >= VOCAB ? VOCAB - 1 : tg);
            ce = -(srow[c] - smax - slse);
        }
        hardP[n] = valid ? ce : 0.f;
        softP[n] = BETA * klt + (1.f - BETA) * kls;
    }
}

// =====================================================================
// Deterministic final reduction over NTOK rows -> scalar loss.
// =====================================================================
__global__ __launch_bounds__(256) void final_kernel(
    const float* __restrict__ hardP, const float* __restrict__ softP,
    const int* __restrict__ tgt, float* __restrict__ out)
{
    __shared__ float redf[256];
    __shared__ int   redi[256];
    const int t = threadIdx.x;

    float hs = 0.f, ss = 0.f;
    int   cnt = 0;
    for (int n = t; n < NTOK; n += 256) {
        hs  += hardP[n];
        ss  += softP[n];
        cnt += (tgt[n] != IGNORE_IDX) ? 1 : 0;
    }

    redf[t] = hs; redi[t] = cnt; __syncthreads();
    for (int s = 128; s > 0; s >>= 1) {
        if (t < s) { redf[t] += redf[t + s]; redi[t] += redi[t + s]; }
        __syncthreads();
    }
    hs  = redf[0];
    cnt = redi[0];
    __syncthreads();

    redf[t] = ss; __syncthreads();
    for (int s = 128; s > 0; s >>= 1) {
        if (t < s) redf[t] += redf[t + s];
        __syncthreads();
    }
    ss = redf[0];

    if (t == 0) {
        const float nvalid    = (cnt > 0) ? (float)cnt : 1.f;
        const float hard_loss = hs / nvalid;
        const float jsd_loss  = ss / (float)NTOK;
        out[0] = W_HARD * hard_loss + W_SOFT * jsd_loss;
    }
}

// =====================================================================
// launcher
// =====================================================================
extern "C" void kernel_launch(void* const* d_in, const int* in_sizes, int n_in,
                              void* d_out, int out_size, void* d_ws, size_t ws_size,
                              hipStream_t stream) {
    (void)in_sizes; (void)n_in; (void)out_size; (void)ws_size;

    const __bf16* student = (const __bf16*)d_in[0];   // [NTOK, HS]
    const __bf16* W_s     = (const __bf16*)d_in[1];   // [VOCAB, HS]
    const __bf16* teacher = (const __bf16*)d_in[2];   // [NTOK, HT]
    const __bf16* W_t     = (const __bf16*)d_in[3];   // [VOCAB, HT]
    const int*    target  = (const int*)d_in[4];      // [NTOK]

    const size_t NV = (size_t)NTOK * (size_t)VOCAB;
    float* sL    = (float*)d_ws;
    float* tL    = sL + NV;
    float* hardP = tL + NV;
    float* softP = hardP + NTOK;

    dim3 block(256);
    dim3 grid((VOCAB + 127) / 128, NTOK / 256);

    gemm_logits_kernel<<<grid, block, 0, stream>>>(student, W_s, sL, HS);
    gemm_logits_kernel<<<grid, block, 0, stream>>>(teacher, W_t, tL, HT);
    row_loss_kernel<<<dim3(NTOK), block, 0, stream>>>(sL, tL, target, hardP, softP);
    final_kernel<<<dim3(1), block, 0, stream>>>(hardP, softP, target, (float*)d_out);
}
